// SwinTransformerBlock_58961311040135
// MI455X (gfx1250) — matrix operand order repaired and checked by reading, not compiled, not run on hardware
//
#include <hip/hip_runtime.h>
#include <hip/hip_bf16.h>

// ---------------- constants ----------------
#define DIMC   192
#define NHC    6
#define HDC    32
#define NTOK   49      // tokens per window (7x7)
#define WSC    7
#define NWIN   64      // windows per image (8x8)
#define HC     56
#define WC     56
#define LTOK   3136    // H*W
#define BATCH  32
#define NROWS  100352  // BATCH * NWIN * NTOK == BATCH * LTOK
#define MLPH   768
#define SHIFTC 3
#define SCALE_F 0.17677669529663687f   // 32^-0.5

typedef _Float16 v16h __attribute__((ext_vector_type(16)));
typedef _Float16 v8h  __attribute__((ext_vector_type(8)));
typedef float    v8f  __attribute__((ext_vector_type(8)));

// ---------------- WMMA fragment loaders (wave32, 16x16x32 f16) ----------------
// A (16xK tile, row-major, stride lda): lanes 0-15 row m0+l, halves 0-15 = K0-15;
// lanes 16-31: K16-31.
__device__ __forceinline__ v16h load_a_frag(const _Float16* A, int lda, int m0,
                                            int k0, int lane) {
  const int r  = lane & 15;
  const int hi = (lane >> 4) & 1;
  const _Float16* p = A + (size_t)(m0 + r) * lda + k0 + hi * 16;
  v8h lo = *(const v8h*)(p);
  v8h hp = *(const v8h*)(p + 8);
  v16h out;
#pragma unroll
  for (int i = 0; i < 8; ++i) { out[i] = lo[i]; out[i + 8] = hp[i]; }
  return out;
}

// B (KxN tile) supplied as B^T row-major [N][K] (== weight [out][in]):
// lane holds column n0+lane%16; lanes 0-15 K k0..k0+15, lanes 16-31 K k0+16..k0+31.
__device__ __forceinline__ v16h load_b_frag(const _Float16* Wt, int ldb, int n0,
                                            int k0, int lane) {
  const int n  = lane & 15;
  const int hi = (lane >> 4) & 1;
  const _Float16* p = Wt + (size_t)(n0 + n) * ldb + k0 + hi * 16;
  return *(const v16h*)(p);
}

#define WMMA_F16(a, b, c) \
  __builtin_amdgcn_wmma_f32_16x16x32_f16(false, (a), false, (b), (short)0, (c), \
                                         false, false)

// 32x32 output tile per wave, double-buffered fragment loads: loads for step
// k+1 are issued before the 4 wmmas of step k so VMEM overlaps matrix math.
template <int K>
__device__ __forceinline__ void gemm_tile_2x2(const _Float16* __restrict__ A,
                                              const _Float16* __restrict__ Wt,
                                              int m0, int n0, int lane,
                                              v8f c[2][2]) {
  v16h a0 = load_a_frag(A, K, m0, 0, lane);
  v16h a1 = load_a_frag(A, K, m0 + 16, 0, lane);
  v16h b0 = load_b_frag(Wt, K, n0, 0, lane);
  v16h b1 = load_b_frag(Wt, K, n0 + 16, 0, lane);
#pragma unroll
  for (int k0 = 0; k0 < K; k0 += 32) {
    v16h na0 = a0, na1 = a1, nb0 = b0, nb1 = b1;
    if (k0 + 32 < K) {
      na0 = load_a_frag(A, K, m0, k0 + 32, lane);
      na1 = load_a_frag(A, K, m0 + 16, k0 + 32, lane);
      nb0 = load_b_frag(Wt, K, n0, k0 + 32, lane);
      nb1 = load_b_frag(Wt, K, n0 + 16, k0 + 32, lane);
    }
    c[0][0] = WMMA_F16(a0, b0, c[0][0]);
    c[0][1] = WMMA_F16(a0, b1, c[0][1]);
    c[1][0] = WMMA_F16(a1, b0, c[1][0]);
    c[1][1] = WMMA_F16(a1, b1, c[1][1]);
    a0 = na0; a1 = na1; b0 = nb0; b1 = nb1;
  }
}

__device__ __forceinline__ float wave_sum(float v) {
#pragma unroll
  for (int off = 16; off > 0; off >>= 1) v += __shfl_xor(v, off, 32);
  return v;
}

// ---------------- weight f32 -> f16 convert ----------------
__global__ __launch_bounds__(256) void cvt_f16_kernel(const float* __restrict__ s,
                                                      _Float16* __restrict__ d,
                                                      int n) {
  int i = blockIdx.x * 256 + threadIdx.x;
  if (i < n) d[i] = (_Float16)s[i];
}

// ---------------- LN1 + roll(-3,-3) + window partition -> f16 ----------------
__global__ __launch_bounds__(256) void ln1_win_kernel(
    const float* __restrict__ x, const float* __restrict__ g,
    const float* __restrict__ b, _Float16* __restrict__ win) {
  const int row  = (blockIdx.x * 256 + threadIdx.x) >> 5;  // window-order row
  const int lane = threadIdx.x & 31;
  if (row >= NROWS) return;
  const int wdw = row / NTOK, pn = row % NTOK;
  const int bb = wdw >> 6, widx = wdw & 63;
  const int wr = widx >> 3, wc = widx & 7;
  const int pr = pn / 7, pc = pn % 7;
  const int ho = (wr * 7 + pr + SHIFTC) % HC;  // rolled coord hs holds x[(hs+3)%56]
  const int wo = (wc * 7 + pc + SHIFTC) % WC;
  const float* src = x + ((size_t)bb * LTOK + ho * WC + wo) * DIMC;
  float v[6];
  float s = 0.f;
#pragma unroll
  for (int i = 0; i < 6; ++i) { v[i] = src[i * 32 + lane]; s += v[i]; }
  s = wave_sum(s);
  const float mean = s * (1.0f / 192.0f);
  float s2 = 0.f;
#pragma unroll
  for (int i = 0; i < 6; ++i) { float d = v[i] - mean; s2 += d * d; }
  s2 = wave_sum(s2);
  const float inv = rsqrtf(s2 * (1.0f / 192.0f) + 1e-5f);
#pragma unroll
  for (int i = 0; i < 6; ++i) {
    int cdx = i * 32 + lane;
    win[(size_t)row * DIMC + cdx] =
        (_Float16)((v[i] - mean) * inv * g[cdx] + b[cdx]);
  }
}

// ---------------- LN2 (plain rows) -> f16 ----------------
__global__ __launch_bounds__(256) void ln2_kernel(const float* __restrict__ x,
                                                  const float* __restrict__ g,
                                                  const float* __restrict__ b,
                                                  _Float16* __restrict__ h2) {
  const int row  = (blockIdx.x * 256 + threadIdx.x) >> 5;
  const int lane = threadIdx.x & 31;
  if (row >= NROWS) return;
  const float* src = x + (size_t)row * DIMC;
  float v[6];
  float s = 0.f;
#pragma unroll
  for (int i = 0; i < 6; ++i) { v[i] = src[i * 32 + lane]; s += v[i]; }
  s = wave_sum(s);
  const float mean = s * (1.0f / 192.0f);
  float s2 = 0.f;
#pragma unroll
  for (int i = 0; i < 6; ++i) { float d = v[i] - mean; s2 += d * d; }
  s2 = wave_sum(s2);
  const float inv = rsqrtf(s2 * (1.0f / 192.0f) + 1e-5f);
#pragma unroll
  for (int i = 0; i < 6; ++i) {
    int cdx = i * 32 + lane;
    h2[(size_t)row * DIMC + cdx] =
        (_Float16)((v[i] - mean) * inv * g[cdx] + b[cdx]);
  }
}

// ---------------- GEMM1: QKV (K=192, N=576) -> scatter q/k/v f16 ----------------
__global__ __launch_bounds__(128) void qkv_gemm_kernel(
    const _Float16* __restrict__ A, const _Float16* __restrict__ Wt,
    const float* __restrict__ bias, _Float16* __restrict__ qb,
    _Float16* __restrict__ kb, _Float16* __restrict__ vb) {
  const int lane = threadIdx.x & 31;
  const int wv = threadIdx.x >> 5;
  const int m0 = blockIdx.x * 64 + (wv >> 1) * 32;
  const int n0 = blockIdx.y * 64 + (wv & 1) * 32;
  v8f c[2][2] = {};
  gemm_tile_2x2<192>(A, Wt, m0, n0, lane, c);
  const int hi = (lane >> 4) & 1;
#pragma unroll
  for (int nt = 0; nt < 2; ++nt) {
    const int n = n0 + nt * 16 + (lane & 15);
    const int which = n / 192;
    const int rem = n % 192;
    const int hh = rem >> 5, d = rem & 31;
    const float bs = bias[n];
#pragma unroll
    for (int mt = 0; mt < 2; ++mt) {
#pragma unroll
      for (int r = 0; r < 8; ++r) {
        const int row = m0 + mt * 16 + r + hi * 8;
        const int w = row / NTOK, pn = row % NTOK;
        const float val = c[mt][nt][r] + bs;
        const size_t o = ((size_t)(w * NHC + hh) * NTOK + pn) * HDC + d;
        if (which == 0)      qb[o] = (_Float16)(val * SCALE_F);
        else if (which == 1) kb[o] = (_Float16)val;
        else                 vb[o] = (_Float16)val;
      }
    }
  }
}

// ---------------- attention: one block per (window, head) ----------------
__global__ __launch_bounds__(128) void attn_kernel(
    const _Float16* __restrict__ qbuf, const _Float16* __restrict__ kbuf,
    const _Float16* __restrict__ vbuf, const float* __restrict__ rpb,
    _Float16* __restrict__ attn_out) {
  __shared__ __align__(32) _Float16 q_lds[64 * 32];
  __shared__ __align__(32) _Float16 k_lds[64 * 32];
  __shared__ __align__(32) _Float16 vt_lds[32 * 64];
  __shared__ __align__(32) float s_lds[64 * 64];
  __shared__ __align__(32) _Float16 p_lds[64 * 64];

  const int wh = blockIdx.x;   // 0..12287
  const int w = wh / NHC;
  const int h = wh % NHC;
  const int widx = w % NWIN;
  const int tid = threadIdx.x;
  const int lane = tid & 31;
  const int wv = tid >> 5;
  const int hi = (lane >> 4) & 1;
  const size_t base = (size_t)wh * (NTOK * HDC);

  for (int idx = tid; idx < 64 * 32; idx += 128) {
    int n = idx >> 5;
    _Float16 qv = (_Float16)0, kv = (_Float16)0;
    if (n < NTOK) { qv = qbuf[base + idx]; kv = kbuf[base + idx]; }
    q_lds[idx] = qv;
    k_lds[idx] = kv;
  }
  for (int idx = tid; idx < 32 * 64; idx += 128) {
    int d = idx >> 6, j = idx & 63;
    vt_lds[idx] = (j < NTOK) ? vbuf[base + j * HDC + d] : (_Float16)0;
  }
  __syncthreads();

  const int m0 = wv * 16;
  const int wr = widx >> 3, wc = widx & 7;

  // S = q k^T (+bias +mask)
  v16h aq = load_a_frag(q_lds, HDC, m0, 0, lane);
#pragma unroll
  for (int nt = 0; nt < 4; ++nt) {
    v16h bk = load_b_frag(k_lds, HDC, nt * 16, 0, lane);
    v8f c = {};
    c = WMMA_F16(aq, bk, c);
    const int n = nt * 16 + (lane & 15);
#pragma unroll
    for (int r = 0; r < 8; ++r) {
      const int m = m0 + r + hi * 8;
      float val = -1e30f;
      if (m < NTOK && n < NTOK) {
        val = c[r];
        const int r1 = m / 7, c1 = m - r1 * 7;
        const int r2 = n / 7, c2 = n - r2 * 7;
        val += rpb[((r1 - r2 + 6) * 13 + (c1 - c2 + 6)) * NHC + h];
        const int hm = wr * 7 + r1, wm = wc * 7 + c1;
        const int hn = wr * 7 + r2, wn = wc * 7 + c2;
        const int rm = (hm < 49 ? 0 : (hm < 53 ? 1 : 2)) * 3 +
                       (wm < 49 ? 0 : (wm < 53 ? 1 : 2));
        const int rn = (hn < 49 ? 0 : (hn < 53 ? 1 : 2)) * 3 +
                       (wn < 49 ? 0 : (wn < 53 ? 1 : 2));
        if (rm != rn) val -= 100.0f;
      }
      s_lds[m * 64 + n] = val;
    }
  }
  __syncthreads();

  // softmax: lanes (l, l+16) share row m0 + (l&15), halves of 64 cols
  {
    const int m = m0 + (lane & 15);
    const int c0 = hi * 32;
    float mx = -3.0e38f;
#pragma unroll
    for (int jj = 0; jj < 32; ++jj) mx = fmaxf(mx, s_lds[m * 64 + c0 + jj]);
    mx = fmaxf(mx, __shfl_xor(mx, 16, 32));
    float sum = 0.f;
#pragma unroll
    for (int jj = 0; jj < 32; ++jj) {
      float e = __expf(s_lds[m * 64 + c0 + jj] - mx);
      s_lds[m * 64 + c0 + jj] = e;
      sum += e;
    }
    sum += __shfl_xor(sum, 16, 32);
    const float inv = 1.0f / sum;
#pragma unroll
    for (int jj = 0; jj < 32; ++jj) {
      const int j = c0 + jj;
      p_lds[m * 64 + j] =
          (j < NTOK) ? (_Float16)(s_lds[m * 64 + j] * inv) : (_Float16)0;
    }
  }
  __syncthreads();

  // O = P v  (K = 64 padded)
#pragma unroll
  for (int dt = 0; dt < 2; ++dt) {
    v8f c = {};
#pragma unroll
    for (int kt = 0; kt < 2; ++kt) {
      v16h ap = load_a_frag(p_lds, 64, m0, kt * 32, lane);
      v16h bv = load_b_frag(vt_lds, 64, dt * 16, kt * 32, lane);
      c = WMMA_F16(ap, bv, c);
    }
    const int d = dt * 16 + (lane & 15);
#pragma unroll
    for (int r = 0; r < 8; ++r) {
      const int m = m0 + r + hi * 8;
      if (m < NTOK)
        attn_out[((size_t)w * NTOK + m) * DIMC + h * HDC + d] = (_Float16)c[r];
    }
  }
}

// ---------------- GEMM2: proj + window reverse + roll(+3) + residual ----------------
__global__ __launch_bounds__(128) void proj_gemm_kernel(
    const _Float16* __restrict__ A, const _Float16* __restrict__ Wt,
    const float* __restrict__ bias, const float* __restrict__ x,
    float* __restrict__ out) {
  const int lane = threadIdx.x & 31;
  const int wv = threadIdx.x >> 5;
  const int m0 = blockIdx.x * 64 + (wv >> 1) * 32;
  const int n0 = blockIdx.y * 64 + (wv & 1) * 32;
  v8f c[2][2] = {};
  gemm_tile_2x2<192>(A, Wt, m0, n0, lane, c);
  const int hi = (lane >> 4) & 1;
#pragma unroll
  for (int nt = 0; nt < 2; ++nt) {
    const int n = n0 + nt * 16 + (lane & 15);
    const float bs = bias[n];
#pragma unroll
    for (int mt = 0; mt < 2; ++mt) {
#pragma unroll
      for (int r = 0; r < 8; ++r) {
        const int row = m0 + mt * 16 + r + hi * 8;
        const int w = row / NTOK, pn = row % NTOK;
        const int bb = w >> 6, widx = w & 63;
        const int wr = widx >> 3, wc = widx & 7;
        const int pr = pn / 7, pc = pn % 7;
        const int ho = (wr * 7 + pr + SHIFTC) % HC;
        const int wo = (wc * 7 + pc + SHIFTC) % WC;
        const size_t o = ((size_t)bb * LTOK + ho * WC + wo) * DIMC + n;
        out[o] = x[o] + c[mt][nt][r] + bs;
      }
    }
  }
}

// ---------------- GEMM3: fc1 + GELU(exact) -> f16 ----------------
__global__ __launch_bounds__(128) void fc1_gemm_kernel(
    const _Float16* __restrict__ A, const _Float16* __restrict__ Wt,
    const float* __restrict__ bias, _Float16* __restrict__ mbuf) {
  const int lane = threadIdx.x & 31;
  const int wv = threadIdx.x >> 5;
  const int m0 = blockIdx.x * 64 + (wv >> 1) * 32;
  const int n0 = blockIdx.y * 64 + (wv & 1) * 32;
  v8f c[2][2] = {};
  gemm_tile_2x2<192>(A, Wt, m0, n0, lane, c);
  const int hi = (lane >> 4) & 1;
#pragma unroll
  for (int nt = 0; nt < 2; ++nt) {
    const int n = n0 + nt * 16 + (lane & 15);
    const float bs = bias[n];
#pragma unroll
    for (int mt = 0; mt < 2; ++mt) {
#pragma unroll
      for (int r = 0; r < 8; ++r) {
        const int row = m0 + mt * 16 + r + hi * 8;
        const float v = c[mt][nt][r] + bs;
        const float ge = 0.5f * v * (1.0f + erff(v * 0.70710678118654752f));
        mbuf[(size_t)row * MLPH + n] = (_Float16)ge;
      }
    }
  }
}

// ---------------- GEMM4: fc2 + residual into d_out ----------------
__global__ __launch_bounds__(128) void fc2_gemm_kernel(
    const _Float16* __restrict__ A, const _Float16* __restrict__ Wt,
    const float* __restrict__ bias, float* __restrict__ out) {
  const int lane = threadIdx.x & 31;
  const int wv = threadIdx.x >> 5;
  const int m0 = blockIdx.x * 64 + (wv >> 1) * 32;
  const int n0 = blockIdx.y * 64 + (wv & 1) * 32;
  v8f c[2][2] = {};
  gemm_tile_2x2<768>(A, Wt, m0, n0, lane, c);
  const int hi = (lane >> 4) & 1;
#pragma unroll
  for (int nt = 0; nt < 2; ++nt) {
    const int n = n0 + nt * 16 + (lane & 15);
    const float bs = bias[n];
#pragma unroll
    for (int mt = 0; mt < 2; ++mt) {
#pragma unroll
      for (int r = 0; r < 8; ++r) {
        const int row = m0 + mt * 16 + r + hi * 8;
        const size_t o = (size_t)row * DIMC + n;
        out[o] = out[o] + c[mt][nt][r] + bs;
      }
    }
  }
}

// ---------------- launch ----------------
extern "C" void kernel_launch(void* const* d_in, const int* in_sizes, int n_in,
                              void* d_out, int out_size, void* d_ws,
                              size_t ws_size, hipStream_t stream) {
  (void)in_sizes; (void)n_in; (void)out_size; (void)ws_size;
  const float* x      = (const float*)d_in[0];
  const float* n1w    = (const float*)d_in[1];
  const float* n1b    = (const float*)d_in[2];
  const float* qkv_w  = (const float*)d_in[3];
  const float* qkv_b  = (const float*)d_in[4];
  const float* rpb    = (const float*)d_in[5];
  const float* proj_w = (const float*)d_in[6];
  const float* proj_b = (const float*)d_in[7];
  const float* n2w    = (const float*)d_in[8];
  const float* n2b    = (const float*)d_in[9];
  const float* fc1_w  = (const float*)d_in[10];
  const float* fc1_b  = (const float*)d_in[11];
  const float* fc2_w  = (const float*)d_in[12];
  const float* fc2_b  = (const float*)d_in[13];
  float* out = (float*)d_out;

  char* ws = (char*)d_ws;
  size_t off = 0;
  auto take = [&](size_t halves) -> _Float16* {
    _Float16* p = (_Float16*)(ws + off);
    off += ((halves * 2 + 255) & ~(size_t)255);
    return p;
  };
  // bufA: win -> attn_out -> h2   (NROWS x 192 f16)
  _Float16* bufA = take((size_t)NROWS * DIMC);
  // bufM: q|k|v overlay, later fc1 output (NROWS x 768 f16)
  _Float16* bufM = take((size_t)NROWS * MLPH);
  _Float16* qb = bufM;
  _Float16* kb = bufM + (size_t)NROWS * DIMC;
  _Float16* vb = bufM + (size_t)2 * NROWS * DIMC;
  _Float16* wqkv16 = take(576 * 192);
  _Float16* wproj16 = take(192 * 192);
  _Float16* wfc1_16 = take((size_t)768 * 192);
  _Float16* wfc2_16 = take((size_t)192 * 768);
  // total workspace ~194 MB

  cvt_f16_kernel<<<(110592 + 255) / 256, 256, 0, stream>>>(qkv_w, wqkv16, 110592);
  cvt_f16_kernel<<<(36864 + 255) / 256, 256, 0, stream>>>(proj_w, wproj16, 36864);
  cvt_f16_kernel<<<(147456 + 255) / 256, 256, 0, stream>>>(fc1_w, wfc1_16, 147456);
  cvt_f16_kernel<<<(147456 + 255) / 256, 256, 0, stream>>>(fc2_w, wfc2_16, 147456);

  ln1_win_kernel<<<NROWS / 8, 256, 0, stream>>>(x, n1w, n1b, bufA);

  // M blocks: 100352/64 = 1568. Each block = 4 waves in 2x2, 64x64 of C.
  qkv_gemm_kernel<<<dim3(1568, 9), 128, 0, stream>>>(bufA, wqkv16, qkv_b, qb, kb, vb);

  attn_kernel<<<2048 * NHC, 128, 0, stream>>>(qb, kb, vb, rpb, bufA);

  proj_gemm_kernel<<<dim3(1568, 3), 128, 0, stream>>>(bufA, wproj16, proj_b, x, out);

  ln2_kernel<<<NROWS / 8, 256, 0, stream>>>(out, n2w, n2b, bufA);

  fc1_gemm_kernel<<<dim3(1568, 12), 128, 0, stream>>>(bufA, wfc1_16, fc1_b, bufM);

  fc2_gemm_kernel<<<dim3(1568, 3), 128, 0, stream>>>(bufM, wfc2_16, fc2_b, out);
}
